// LayoutLMv3TransformerLayer_62852551409734
// MI455X (gfx1250) — compile-verified, matrix operand order checked by reading, not run on hardware
//
#include <hip/hip_runtime.h>
#include <cstdint>

// ---------------- problem constants (from the reference) ----------------
#define BB   4
#define SS   2048
#define DD   768
#define HH   12
#define DHH  64
#define II   3072
#define MM   (BB*SS)   // 8192 tokens
#define RBINS 32

typedef __bf16 bf16;
typedef __bf16 v16bf __attribute__((ext_vector_type(16)));
typedef float  v8f   __attribute__((ext_vector_type(8)));

static __device__ __forceinline__ bf16 f2bf(float f) {
  unsigned u = __float_as_uint(f);
  u += 0x7FFFu + ((u >> 16) & 1u);            // round-to-nearest-even
  unsigned short h = (unsigned short)(u >> 16);
  bf16 r; __builtin_memcpy(&r, &h, 2); return r;
}

union FragU { v16bf v; uint4 q[2]; };

// Load one 16x32 bf16 WMMA operand fragment (A-style; B uses the transposed
// matrix with identical addressing). Per the CDNA5 16-bit operand layout:
// lane L: row = L%16, K = {half*8 .. half*8+7} and {half*8+16 .. half*8+23}.
static __device__ __forceinline__ v16bf load_frag(const bf16* __restrict__ base, int ld) {
  const int lane = threadIdx.x & 31;
  const bf16* p = base + (size_t)(lane & 15) * ld + ((lane >> 4) << 3);
  FragU f;
  f.q[0] = *(const uint4*)(p);
  f.q[1] = *(const uint4*)(p + 16);
  return f.v;
}

static __device__ __forceinline__ v8f wmma_bf16(v16bf a, v16bf b, v8f c) {
  // D = A(16x32) * B(32x16) + C, f32 accumulate
  return __builtin_amdgcn_wmma_f32_16x16x32_bf16(false, a, false, b, (short)0, c, false, false);
}

// CDNA5 async global->LDS copy (16B per lane), tracked by ASYNCcnt.
static __device__ __forceinline__ void async_b128_to_lds(unsigned lds_off, const bf16* g) {
  asm volatile("global_load_async_to_lds_b128 %0, %1, off"
               :: "v"(lds_off), "v"(g) : "memory");
}
static __device__ __forceinline__ void wait_async0() {
  asm volatile("s_wait_asynccnt 0x0" ::: "memory");
}

// ---------------- small prep kernels ----------------
__global__ __launch_bounds__(256)
void cast_f32_bf16(const float* __restrict__ in, bf16* __restrict__ out, int n) {
  int i = blockIdx.x * 256 + threadIdx.x;
  if (i < n) out[i] = f2bf(in[i]);
}

// W[K,N] f32 -> Wt[N,K] bf16
__global__ __launch_bounds__(256)
void transpose_cast(const float* __restrict__ W, bf16* __restrict__ Wt, int K, int N) {
  int i = blockIdx.x * 256 + threadIdx.x;
  if (i < K * N) {
    int k = i / N, n = i % N;
    Wt[(size_t)n * K + k] = f2bf(W[i]);
  }
}

// ---------------- WMMA GEMM: C[M,N] = A[M,K] * Bt[N,K]^T + bias ----------------
// Block: 128 threads = 4 waves (2x2). Block tile 128x128, wave tile 64x64
// (4x4 WMMA accumulators = 16 v_wmma per 32-deep k-step).
// A tile (128x32 bf16 = 8KB) is staged global->LDS with CDNA5 async copies,
// double buffered so the k+32 prefetch overlaps the 16 WMMAs of step k.
#define EP_BF16 0
#define EP_F32  1
#define EP_GELU 2
#define EP_Q    3
#define EP_K    4
#define EP_V    5
#define KT 32

__global__ __launch_bounds__(128)
void gemm_bf16_wmma(const bf16* __restrict__ A, const bf16* __restrict__ Bt,
                    const float* __restrict__ bias, void* __restrict__ Cout,
                    int M, int N, int K, int mode)
{
  __shared__ __align__(16) bf16 As[2][128 * KT];   // 2 x 8KB double buffer

  const int tid  = threadIdx.x;
  const int wave = tid >> 5;
  const int lane = tid & 31;
  const int wm = wave >> 1, wn = wave & 1;
  const int blockRow = blockIdx.y * 128;
  const int rowLds   = wm * 64;                      // wave's rows inside LDS tile
  const int colBase  = blockIdx.x * 128 + wn * 64;

  v8f acc[4][4];
  for (int i = 0; i < 4; ++i)
    for (int j = 0; j < 4; ++j)
      for (int e = 0; e < 8; ++e) acc[i][j][e] = 0.f;

  // cooperative async stage of the 128x32 A tile: 512 x 16B chunks / 128 threads
  auto stage = [&](int buf, int k) {
#pragma unroll
    for (int i = 0; i < 4; ++i) {
      int c = i * 128 + tid;                 // 0..511
      int r = c >> 2;                        // row 0..127
      int o = (c & 3) * 8;                   // 8 bf16 per 16B chunk
      unsigned lds = (unsigned)(uintptr_t)&As[buf][r * KT + o];
      async_b128_to_lds(lds, A + (size_t)(blockRow + r) * K + k + o);
    }
  };

  stage(0, 0);
  wait_async0();
  __syncthreads();

  int buf = 0;
  for (int k = 0; k < K; k += KT) {
    const int kn = k + KT;
    if (kn < K) {
      stage(buf ^ 1, kn);                    // async prefetch next A tile
#pragma unroll
      for (int nt = 0; nt < 4; ++nt)         // prefetch next B fragments into L2/L0
        __builtin_prefetch(Bt + (size_t)(colBase + nt * 16 + (lane & 15)) * K + kn, 0, 0);
    }

    v16bf a[4], b[4];
#pragma unroll
    for (int nt = 0; nt < 4; ++nt)
      b[nt] = load_frag(Bt + (size_t)(colBase + nt * 16) * K + k, K);
#pragma unroll
    for (int mt = 0; mt < 4; ++mt)
      a[mt] = load_frag(&As[buf][(rowLds + mt * 16) * KT], KT);

#pragma unroll
    for (int mt = 0; mt < 4; ++mt)
#pragma unroll
      for (int nt = 0; nt < 4; ++nt)
        acc[mt][nt] = wmma_bf16(a[mt], b[nt], acc[mt][nt]);

    wait_async0();                           // prefetch landed
    __syncthreads();                         // everyone done reading As[buf]
    buf ^= 1;
  }

  const int half = lane >> 4, cl = lane & 15;
#pragma unroll
  for (int mt = 0; mt < 4; ++mt)
#pragma unroll
    for (int nt = 0; nt < 4; ++nt)
#pragma unroll
      for (int e = 0; e < 8; ++e) {
        int row = blockRow + rowLds + mt * 16 + e + 8 * half;
        int col = colBase + nt * 16 + cl;
        float v = acc[mt][nt][e] + bias[col];
        if (mode == EP_BF16) {
          ((bf16*)Cout)[(size_t)row * N + col] = f2bf(v);
        } else if (mode == EP_F32) {
          ((float*)Cout)[(size_t)row * N + col] = v;
        } else if (mode == EP_GELU) {
          float g = 0.5f * v * (1.0f + erff(v * 0.70710678118654752f));
          ((bf16*)Cout)[(size_t)row * N + col] = f2bf(g);
        } else {
          int bI = row / SS, s = row % SS;
          int h  = col / DHH, d = col % DHH;
          if (mode == EP_Q) {        // [BH, S, DH], pre-scaled by 1/sqrt(DH)
            ((bf16*)Cout)[(((size_t)(bI * HH + h) * SS + s) << 6) + d] = f2bf(v * 0.125f);
          } else if (mode == EP_K) { // [BH, S, DH]
            ((bf16*)Cout)[(((size_t)(bI * HH + h) * SS + s) << 6) + d] = f2bf(v);
          } else {                   // EP_V: transposed [BH, DH, S]
            ((bf16*)Cout)[((size_t)(bI * HH + h) * DHH + d) * SS + s] = f2bf(v);
          }
        }
      }
}

// ---------------- flash attention (one wave per 16-row q tile) ----------------
__global__ __launch_bounds__(128)
void attn_fa_wmma(const bf16* __restrict__ Q, const bf16* __restrict__ Km,
                  const bf16* __restrict__ Vt, const float* __restrict__ rel_bias,
                  const float* __restrict__ mask, bf16* __restrict__ ctx)
{
  __shared__ float sc[4][16 * 32];      // score tile per wave
  __shared__ __align__(16) bf16 pp[4][16 * 32];   // probs (bf16) per wave
  __shared__ float salpha[4][16];
  __shared__ float sl[4][16];

  const int w    = threadIdx.x >> 5;
  const int lane = threadIdx.x & 31;
  const int half = lane >> 4, cl = lane & 15;

  const int tile   = blockIdx.x * 4 + w;      // global wave tile id
  const int qtiles = SS / 16;                 // 128
  const int bh     = tile / qtiles;
  const int qbase  = (tile % qtiles) * 16;
  const int bI = bh / HH, h = bh % HH;

  const bf16* Qb = Q + (size_t)bh * SS * DHH + (size_t)qbase * DHH;
  v16bf qa0 = load_frag(Qb, DHH);             // K-dim 0..31
  v16bf qa1 = load_frag(Qb + 32, DHH);        // K-dim 32..63

  v8f o[4];
  for (int t = 0; t < 4; ++t) for (int e = 0; e < 8; ++e) o[t][e] = 0.f;
  float m_i = -1e30f, l_i = 0.f;

  for (int kb = 0; kb < SS; kb += 32) {
    // scores S[16 x 32] = Q * K^T (Q already scaled)
    v8f s[2];
    for (int t = 0; t < 2; ++t) for (int e = 0; e < 8; ++e) s[t][e] = 0.f;
#pragma unroll
    for (int nt = 0; nt < 2; ++nt) {
      const bf16* Kb = Km + (size_t)bh * SS * DHH + (size_t)(kb + nt * 16) * DHH;
      s[nt] = wmma_bf16(qa0, load_frag(Kb, DHH), s[nt]);
      s[nt] = wmma_bf16(qa1, load_frag(Kb + 32, DHH), s[nt]);
    }
    // epilogue: relative-position bias + mask, spill to LDS
#pragma unroll
    for (int nt = 0; nt < 2; ++nt)
#pragma unroll
      for (int e = 0; e < 8; ++e) {
        int qr = qbase + e + 8 * half;
        int kp = kb + nt * 16 + cl;
        int rel = qr - kp + (RBINS - 1);
        rel = rel < 0 ? 0 : (rel > 2 * RBINS - 2 ? 2 * RBINS - 2 : rel);
        float v = s[nt][e] + rel_bias[rel * HH + h]
                + (1.0f - mask[bI * SS + kp]) * -10000.0f;
        sc[w][(e + 8 * half) * 32 + nt * 16 + cl] = v;
      }
    __syncthreads();

    // online softmax: lanes 0..15 each own one row
    if (lane < 16) {
      const float* r = &sc[w][lane * 32];
      float mx = m_i;
      for (int j = 0; j < 32; ++j) mx = fmaxf(mx, r[j]);
      float alpha = __expf(m_i - mx);
      float sum = 0.f;
      for (int j = 0; j < 32; ++j) {
        float p = __expf(r[j] - mx);
        pp[w][lane * 32 + j] = f2bf(p);
        sum += p;
      }
      l_i = l_i * alpha + sum;
      m_i = mx;
      salpha[w][lane] = alpha;
    }
    __syncthreads();

    // rescale running O, then O += P(16x32) * V(32x64)
#pragma unroll
    for (int t = 0; t < 4; ++t)
#pragma unroll
      for (int e = 0; e < 8; ++e) o[t][e] *= salpha[w][e + 8 * half];

    v16bf pa = load_frag(&pp[w][0], 32);      // probs reloaded from LDS as A frag
#pragma unroll
    for (int t = 0; t < 4; ++t) {
      const bf16* Vb = Vt + ((size_t)bh * DHH + t * 16) * SS + kb;  // Vt: [BH, DH, S]
      o[t] = wmma_bf16(pa, load_frag(Vb, SS), o[t]);
    }
    // next iteration's sc writes are ordered by its own first barrier
  }

  if (lane < 16) sl[w][lane] = l_i;
  __syncthreads();
#pragma unroll
  for (int t = 0; t < 4; ++t)
#pragma unroll
    for (int e = 0; e < 8; ++e) {
      int row = qbase + e + 8 * half;
      float inv = 1.0f / (sl[w][e + 8 * half] + 1e-20f);
      int col = t * 16 + cl;
      ctx[((size_t)(bI * SS + row)) * DD + h * DHH + col] = f2bf(o[t][e] * inv);
    }
}

// ---------------- fused residual-add + LayerNorm ----------------
__global__ __launch_bounds__(256)
void add_ln(const float* __restrict__ a, const float* __restrict__ b,
            const float* __restrict__ g, const float* __restrict__ be,
            float* __restrict__ outf, bf16* __restrict__ outb)
{
  __shared__ float red[256];
  const int row = blockIdx.x;
  const int tid = threadIdx.x;
  const float* pa = a + (size_t)row * DD;
  const float* pb = b + (size_t)row * DD;

  float v[3]; float s = 0.f;
#pragma unroll
  for (int i = 0; i < 3; ++i) { v[i] = pa[tid + i * 256] + pb[tid + i * 256]; s += v[i]; }
  red[tid] = s; __syncthreads();
  for (int o = 128; o > 0; o >>= 1) { if (tid < o) red[tid] += red[tid + o]; __syncthreads(); }
  float mean = red[0] * (1.0f / DD);
  __syncthreads();

  float s2 = 0.f;
#pragma unroll
  for (int i = 0; i < 3; ++i) { float d = v[i] - mean; s2 += d * d; }
  red[tid] = s2; __syncthreads();
  for (int o = 128; o > 0; o >>= 1) { if (tid < o) red[tid] += red[tid + o]; __syncthreads(); }
  float rstd = rsqrtf(red[0] * (1.0f / DD) + 1e-12f);

#pragma unroll
  for (int i = 0; i < 3; ++i) {
    int c = tid + i * 256;
    float o = (v[i] - mean) * rstd * g[c] + be[c];
    if (outf) outf[(size_t)row * DD + c] = o;
    if (outb) outb[(size_t)row * DD + c] = f2bf(o);
  }
}

// ---------------- host orchestration ----------------
extern "C" void kernel_launch(void* const* d_in, const int* in_sizes, int n_in,
                              void* d_out, int out_size, void* d_ws, size_t ws_size,
                              hipStream_t stream)
{
  (void)in_sizes; (void)n_in; (void)out_size; (void)ws_size;
  const float* x     = (const float*)d_in[0];
  const float* mask  = (const float*)d_in[1];
  const float* wq    = (const float*)d_in[2];
  const float* bq    = (const float*)d_in[3];
  const float* wk    = (const float*)d_in[4];
  const float* bk    = (const float*)d_in[5];
  const float* wv    = (const float*)d_in[6];
  const float* bv    = (const float*)d_in[7];
  const float* wo    = (const float*)d_in[8];
  const float* bo    = (const float*)d_in[9];
  const float* relb  = (const float*)d_in[10];
  const float* w_ao  = (const float*)d_in[11];
  const float* b_ao  = (const float*)d_in[12];
  const float* ln1g  = (const float*)d_in[13];
  const float* ln1b  = (const float*)d_in[14];
  const float* w_i   = (const float*)d_in[15];
  const float* b_i   = (const float*)d_in[16];
  const float* w_out = (const float*)d_in[17];
  const float* b_out = (const float*)d_in[18];
  const float* ln2g  = (const float*)d_in[19];
  const float* ln2b  = (const float*)d_in[20];

  char* ws = (char*)d_ws;
  size_t off = 0;
  auto take = [&](size_t bytes) -> char* {
    char* p = ws + off;
    off = (off + bytes + 255) & ~(size_t)255;
    return p;
  };

  const size_t actB  = (size_t)MM * DD * sizeof(bf16);     // 12.58 MB
  const size_t actF  = (size_t)MM * DD * sizeof(float);    // 25.17 MB
  const size_t hdB   = (size_t)BB * HH * SS * DHH * sizeof(bf16); // == actB

  bf16* xb     = (bf16*)take(actB);
  bf16* wqt    = (bf16*)take((size_t)DD * DD * 2);
  bf16* wkt    = (bf16*)take((size_t)DD * DD * 2);
  bf16* wvt    = (bf16*)take((size_t)DD * DD * 2);
  bf16* wot    = (bf16*)take((size_t)DD * DD * 2);
  bf16* waot   = (bf16*)take((size_t)DD * DD * 2);
  bf16* wit    = (bf16*)take((size_t)II * DD * 2);   // w_i^T  [I,D]
  bf16* woutt  = (bf16*)take((size_t)DD * II * 2);   // w_out^T [D,I]
  char* qkv0   = ws + off;                           // Q/K/Vt/ctx block (aliased by inter)
  bf16* Qb     = (bf16*)take(hdB);
  bf16* Kb     = (bf16*)take(hdB);
  bf16* Vt     = (bf16*)take(hdB);
  bf16* ctx    = (bf16*)take(actB);
  bf16* attnb  = (bf16*)take(actB);
  float* ao    = (float*)take(actF);
  float* h1f   = (float*)take(actF);
  bf16* h1b    = (bf16*)take(actB);
  bf16* inter  = (bf16*)qkv0;        // 50.33 MB, exactly reuses Q/K/Vt/ctx (all dead)
  float* out2  = ao;                 // ao dead after LN1

  // 1) weight transpose+cast, activation cast
  auto tc = [&](const float* W, bf16* Wt, int K, int N) {
    int n = K * N;
    transpose_cast<<<(n + 255) / 256, 256, 0, stream>>>(W, Wt, K, N);
  };
  tc(wq, wqt, DD, DD);
  tc(wk, wkt, DD, DD);
  tc(wv, wvt, DD, DD);
  tc(wo, wot, DD, DD);
  tc(w_ao, waot, DD, DD);
  tc(w_i, wit, DD, II);
  tc(w_out, woutt, II, DD);
  cast_f32_bf16<<<(MM * DD + 255) / 256, 256, 0, stream>>>(x, xb, MM * DD);

  // 2) QKV projections (fused scale + head layout)
  dim3 gD(DD / 128, MM / 128);
  gemm_bf16_wmma<<<gD, 128, 0, stream>>>(xb, wqt, bq, Qb, MM, DD, DD, EP_Q);
  gemm_bf16_wmma<<<gD, 128, 0, stream>>>(xb, wkt, bk, Kb, MM, DD, DD, EP_K);
  gemm_bf16_wmma<<<gD, 128, 0, stream>>>(xb, wvt, bv, Vt, MM, DD, DD, EP_V);

  // 3) flash attention -> ctx [M, D]
  attn_fa_wmma<<<(BB * HH * (SS / 16)) / 4, 128, 0, stream>>>(Qb, Kb, Vt, relb, mask, ctx);

  // 4) output projection chain
  gemm_bf16_wmma<<<gD, 128, 0, stream>>>(ctx, wot, bo, attnb, MM, DD, DD, EP_BF16);
  gemm_bf16_wmma<<<gD, 128, 0, stream>>>(attnb, waot, b_ao, ao, MM, DD, DD, EP_F32);

  // 5) LN1: h1 = LN(ao + x)
  add_ln<<<MM, 256, 0, stream>>>(ao, x, ln1g, ln1b, h1f, h1b);

  // 6) FFN
  dim3 gI(II / 128, MM / 128);
  gemm_bf16_wmma<<<gI, 128, 0, stream>>>(h1b, wit, b_i, inter, MM, II, DD, EP_GELU);
  gemm_bf16_wmma<<<gD, 128, 0, stream>>>(inter, woutt, b_out, out2, MM, DD, II, EP_F32);

  // 7) LN2 -> d_out (f32)
  add_ln<<<MM, 256, 0, stream>>>(out2, h1f, ln2g, ln2b, (float*)d_out, nullptr);
}
